// RWKV7Attention_19662360281591
// MI455X (gfx1250) — matrix operand order repaired and checked by reading, not compiled, not run on hardware
//
#include <hip/hip_runtime.h>
#include <math.h>

#define DEVINL __device__ __forceinline__

typedef unsigned short u16;
typedef __attribute__((ext_vector_type(8)))  __bf16 bf16x8;
typedef __attribute__((ext_vector_type(16))) __bf16 bf16x16;
typedef __attribute__((ext_vector_type(8)))  float  f32x8;

static constexpr int BATCH  = 2;
static constexpr int TLEN   = 2048;
static constexpr int DMODEL = 1024;
static constexpr int MROWS  = BATCH * TLEN;      // 4096
static constexpr int NHEADS = 16;
static constexpr int HKD    = 64;
static constexpr float W_SCALE_C = -0.6065306597126334f;
static constexpr float GN_EPS_C  = 64.0f * 1e-5f;

// ---------------------------------------------------------------- helpers ---
DEVINL u16 f2bf(float f) {
  unsigned u = __float_as_uint(f);
  u += 0x7FFFu + ((u >> 16) & 1u);               // round-to-nearest-even
  return (u16)(u >> 16);
}
DEVINL float sigmoidf_(float x) { return 1.0f / (1.0f + __expf(-x)); }

DEVINL bf16x16 combine16(bf16x8 lo, bf16x8 hi) {
  bf16x16 r;
#pragma unroll
  for (int i = 0; i < 8; ++i) { r[i] = lo[i]; r[i + 8] = hi[i]; }
  return r;
}

// A fragment (16x32, M x K): lane l -> row m0+(l&15); K halves {ks..ks+7, 16+ks..}
DEVINL bf16x16 load_fragA(const u16* __restrict__ A, int lda, int m0, int k0,
                          int lane, int mclamp) {
  int m = m0 + (lane & 15);
  m = (m < mclamp) ? m : mclamp;
  int ks = (lane >> 4) * 8;
  const __bf16* p = reinterpret_cast<const __bf16*>(A) + (size_t)m * lda + k0 + ks;
  bf16x8 lo = *reinterpret_cast<const bf16x8*>(p);
  bf16x8 hi = *reinterpret_cast<const bf16x8*>(p + 16);
  return combine16(lo, hi);
}

// B fragment (32x16, K x N) from row-major W[N,K]: lane l -> col n0+(l&15),
// K halves {ks..ks+7, 16+ks..}, ks = (l>>4)*8.
DEVINL bf16x16 load_fragB(const u16* __restrict__ W, int ldb, int n0, int k0,
                          int lane, int nclamp) {
  int n = n0 + (lane & 15);
  n = (n < nclamp) ? n : nclamp;
  int ks = (lane >> 4) * 8;
  const __bf16* p = reinterpret_cast<const __bf16*>(W) + (size_t)n * ldb + k0 + ks;
  bf16x8 lo = *reinterpret_cast<const bf16x8*>(p);
  bf16x8 hi = *reinterpret_cast<const bf16x8*>(p + 16);
  return combine16(lo, hi);
}

// ------------------------------------------------------------ conversion ---
__global__ __launch_bounds__(256)
void conv_f32_bf16_kernel(const float* __restrict__ src, u16* __restrict__ dst,
                          int rows, int cols, int stride) {
  int idx = blockIdx.x * 256 + threadIdx.x;
  if (idx >= rows * stride) return;
  int r = idx / stride, c = idx - r * stride;
  dst[idx] = (c < cols) ? f2bf(src[(size_t)r * cols + c]) : (u16)0;
}

// ---------------------------------------------------------------- mixing ---
__global__ __launch_bounds__(256)
void mix_kernel(const float* __restrict__ x,
                const float* __restrict__ cr, const float* __restrict__ cw,
                const float* __restrict__ ck, const float* __restrict__ cv,
                const float* __restrict__ ca, const float* __restrict__ cg,
                u16* __restrict__ oR, u16* __restrict__ oW, u16* __restrict__ oK,
                u16* __restrict__ oV, u16* __restrict__ oA, u16* __restrict__ oG) {
  int idx = blockIdx.x * 256 + threadIdx.x;
  if (idx >= MROWS * DMODEL) return;
  int d = idx & (DMODEL - 1);
  int row = idx >> 10;
  int t = row & (TLEN - 1);
  float xc = x[idx];
  float sh = (t == 0) ? 0.0f : x[idx - DMODEL];
  float delta = sh - xc;
  oR[idx] = f2bf(xc + delta * cr[d]);
  oW[idx] = f2bf(xc + delta * cw[d]);
  oK[idx] = f2bf(xc + delta * ck[d]);
  oV[idx] = f2bf(xc + delta * cv[d]);
  oA[idx] = f2bf(xc + delta * ca[d]);
  oG[idx] = f2bf(xc + delta * cg[d]);
}

// -------------------------------------------------------------- WMMA GEMM ---
// C[M,N] = A[M,K] @ W[N,K]^T.  A,W bf16 row-major.  8 waves -> 64x256 block tile,
// each wave a 32x64 tile = 8 v_wmma_f32_16x16x32_bf16 accumulators.
// Register double-buffered: loads for k+32 are issued before the WMMA block of k.
// mode: 0 f32 | 1 bf16 | 2 tanh->bf16 | 3 sigmoid->bf16
//       4 sigmoid(x+bias)->f32 | 5 scale*sigmoid(x+bias)->f32
__global__ __launch_bounds__(256)
void gemm_bf16_wmma_kernel(const u16* __restrict__ A, const u16* __restrict__ W,
                           const float* __restrict__ bias,
                           float* __restrict__ outF, u16* __restrict__ outH,
                           int M, int N, int K, int lda, int ldb, int ldc,
                           int Npad, int mode, float scale) {
  const int lane = threadIdx.x & 31;
  const int wid  = threadIdx.x >> 5;
  const int wm = wid & 1, wn = wid >> 1;
  const int m0 = blockIdx.y * 64 + wm * 32;
  const int n0 = blockIdx.x * 256 + wn * 64;
  if (m0 >= M || n0 >= Npad) return;

  f32x8 acc[2][4];
#pragma unroll
  for (int i = 0; i < 2; ++i)
#pragma unroll
    for (int j = 0; j < 4; ++j) acc[i][j] = (f32x8){};

  // ---- prologue: fragments for k0 = 0
  bf16x16 a0 = load_fragA(A, lda, m0,      0, lane, M - 1);
  bf16x16 a1 = load_fragA(A, lda, m0 + 16, 0, lane, M - 1);
  bf16x16 b0 = load_fragB(W, ldb, n0,      0, lane, N - 1);
  bf16x16 b1 = load_fragB(W, ldb, n0 + 16, 0, lane, N - 1);
  bf16x16 b2 = load_fragB(W, ldb, n0 + 32, 0, lane, N - 1);
  bf16x16 b3 = load_fragB(W, ldb, n0 + 48, 0, lane, N - 1);

  for (int k0 = 0; k0 < K; k0 += 32) {
    const int kn = k0 + 32;
    // long-range prefetch (two iterations ahead) into L2/L1
    if (kn + 32 < K)
      __builtin_prefetch(A + (size_t)(m0 + (lane & 15)) * lda + kn + 32, 0, 3);
    // ---- issue next iteration's fragment loads before consuming current
    bf16x16 na0 = a0, na1 = a1, nb0 = b0, nb1 = b1, nb2 = b2, nb3 = b3;
    if (kn < K) {
      na0 = load_fragA(A, lda, m0,      kn, lane, M - 1);
      na1 = load_fragA(A, lda, m0 + 16, kn, lane, M - 1);
      nb0 = load_fragB(W, ldb, n0,      kn, lane, N - 1);
      nb1 = load_fragB(W, ldb, n0 + 16, kn, lane, N - 1);
      nb2 = load_fragB(W, ldb, n0 + 32, kn, lane, N - 1);
      nb3 = load_fragB(W, ldb, n0 + 48, kn, lane, N - 1);
    }
    // ---- 8 WMMAs on current fragments (hides the loads above)
    acc[0][0] = __builtin_amdgcn_wmma_f32_16x16x32_bf16(false, a0, false, b0, (short)0, acc[0][0], false, false);
    acc[1][0] = __builtin_amdgcn_wmma_f32_16x16x32_bf16(false, a1, false, b0, (short)0, acc[1][0], false, false);
    acc[0][1] = __builtin_amdgcn_wmma_f32_16x16x32_bf16(false, a0, false, b1, (short)0, acc[0][1], false, false);
    acc[1][1] = __builtin_amdgcn_wmma_f32_16x16x32_bf16(false, a1, false, b1, (short)0, acc[1][1], false, false);
    acc[0][2] = __builtin_amdgcn_wmma_f32_16x16x32_bf16(false, a0, false, b2, (short)0, acc[0][2], false, false);
    acc[1][2] = __builtin_amdgcn_wmma_f32_16x16x32_bf16(false, a1, false, b2, (short)0, acc[1][2], false, false);
    acc[0][3] = __builtin_amdgcn_wmma_f32_16x16x32_bf16(false, a0, false, b3, (short)0, acc[0][3], false, false);
    acc[1][3] = __builtin_amdgcn_wmma_f32_16x16x32_bf16(false, a1, false, b3, (short)0, acc[1][3], false, false);
    a0 = na0; a1 = na1; b0 = nb0; b1 = nb1; b2 = nb2; b3 = nb3;
  }

  const int crow = (lane >> 4) * 8;   // C layout: VGPR i -> row i + 8*(lane>=16)
  const int ccol = lane & 15;
#pragma unroll
  for (int mi = 0; mi < 2; ++mi) {
#pragma unroll
    for (int j = 0; j < 4; ++j) {
      int n = n0 + j * 16 + ccol;
      if (n >= Npad) continue;
      bool inN = (n < N);
      float bv = (bias != nullptr && inN) ? bias[n] : 0.0f;
#pragma unroll
      for (int ri = 0; ri < 8; ++ri) {
        int m = m0 + mi * 16 + crow + ri;
        if (m >= M) continue;
        float y = inN ? acc[mi][j][ri] : 0.0f;
        size_t oi = (size_t)m * ldc + n;
        switch (mode) {
          case 0: outF[oi] = y; break;
          case 1: outH[oi] = f2bf(y); break;
          case 2: outH[oi] = f2bf(tanhf(y)); break;
          case 3: outH[oi] = f2bf(sigmoidf_(y)); break;
          case 4: outF[oi] = sigmoidf_(y + bv); break;
          default: outF[oi] = scale * sigmoidf_(y + bv); break;
        }
      }
    }
  }
}

// ------------------------------------------------------ pre-scan elementwise ---
// One wave32 per (token, head); lane handles channels {lane, lane+32}.
__global__ __launch_bounds__(256)
void prep_kernel(float* __restrict__ kbuf,      // in: k_raw, out: k updated
                 float* __restrict__ abuf,      // in: a(sigmoid), out: avec=-kk
                 float* __restrict__ vbuf,      // in: v_raw, out: lerped v
                 const float* __restrict__ vmix,
                 const float* __restrict__ v_first,
                 const float* __restrict__ k_k, const float* __restrict__ k_a,
                 float* __restrict__ bvec) {
  int gw = (blockIdx.x * 256 + threadIdx.x) >> 5;
  int lane = threadIdx.x & 31;
  if (gw >= MROWS * NHEADS) return;
  int bt = gw >> 4, h = gw & 15;
  size_t base = (size_t)bt * DMODEL + h * HKD;
  int c0 = lane, c1 = lane + 32;
  float kr0 = kbuf[base + c0], kr1 = kbuf[base + c1];
  float a0 = abuf[base + c0],  a1 = abuf[base + c1];
  float kk0 = kr0 * k_k[h * HKD + c0];
  float kk1 = kr1 * k_k[h * HKD + c1];
  float ss = kk0 * kk0 + kk1 * kk1;
#pragma unroll
  for (int off = 16; off >= 1; off >>= 1) ss += __shfl_xor(ss, off, 32);
  float invn = 1.0f / fmaxf(sqrtf(ss), 1e-12f);
  kk0 *= invn; kk1 *= invn;
  bvec[base + c0] = kk0 * a0;
  bvec[base + c1] = kk1 * a1;
  float vr0 = vbuf[base + c0], vr1 = vbuf[base + c1];
  float vm0 = vmix[base + c0], vm1 = vmix[base + c1];
  vbuf[base + c0] = vr0 + vm0 * (v_first[base + c0] - vr0);
  vbuf[base + c1] = vr1 + vm1 * (v_first[base + c1] - vr1);
  kbuf[base + c0] = kr0 * (1.0f + (a0 - 1.0f) * k_a[h * HKD + c0]);
  kbuf[base + c1] = kr1 * (1.0f + (a1 - 1.0f) * k_a[h * HKD + c1]);
  abuf[base + c0] = -kk0;
  abuf[base + c1] = -kk1;
}

// --------------------------------------------------------------- recurrence ---
// One block of 64 threads per (batch, head); thread t owns state row V=t (64 regs).
__global__ __launch_bounds__(64)
void rwkv7_scan_kernel(const float* __restrict__ r, const float* __restrict__ w,
                       const float* __restrict__ k, const float* __restrict__ v,
                       const float* __restrict__ avec, const float* __restrict__ bvec,
                       float* __restrict__ o) {
  const int bh = blockIdx.x;
  const int b = bh >> 4, h = bh & 15;
  const int tid = threadIdx.x;
  __shared__ float s_ew[HKD], s_a[HKD], s_b[HKD], s_k[HKD], s_r[HKD];
  float S[HKD];
#pragma unroll
  for (int j = 0; j < HKD; ++j) S[j] = 0.0f;
  size_t base = ((size_t)b * TLEN) * DMODEL + (size_t)h * HKD + tid;
  for (int t = 0; t < TLEN; ++t, base += DMODEL) {
    s_ew[tid] = __expf(w[base]);
    s_a[tid]  = avec[base];
    s_b[tid]  = bvec[base];
    s_k[tid]  = k[base];
    s_r[tid]  = r[base];
    float vv = v[base];
    __syncthreads();
    float Sa = 0.0f;
#pragma unroll
    for (int j = 0; j < HKD; ++j) Sa += S[j] * s_a[j];
    float out = 0.0f;
#pragma unroll
    for (int j = 0; j < HKD; ++j) {
      float s = S[j] * s_ew[j] + Sa * s_b[j] + vv * s_k[j];
      S[j] = s;
      out += s * s_r[j];
    }
    o[base] = out;
    __syncthreads();
  }
}

// ------------------------------------------------------ groupnorm + bonus ---
__global__ __launch_bounds__(256)
void gn_bonus_kernel(const float* __restrict__ o, const float* __restrict__ r,
                     const float* __restrict__ k, const float* __restrict__ v,
                     const float* __restrict__ r_k,
                     const float* __restrict__ gamma, const float* __restrict__ beta,
                     const float* __restrict__ g, u16* __restrict__ y16) {
  int gw = (blockIdx.x * 256 + threadIdx.x) >> 5;
  int lane = threadIdx.x & 31;
  if (gw >= MROWS * NHEADS) return;
  int bt = gw >> 4, h = gw & 15;
  size_t base = (size_t)bt * DMODEL + h * HKD;
  int c0 = lane, c1 = lane + 32;
  float o0 = o[base + c0], o1 = o[base + c1];
  float s = o0 + o1;
  float sq = o0 * o0 + o1 * o1;
  float dp = r[base + c0] * k[base + c0] * r_k[h * HKD + c0] +
             r[base + c1] * k[base + c1] * r_k[h * HKD + c1];
#pragma unroll
  for (int off = 16; off >= 1; off >>= 1) {
    s  += __shfl_xor(s, off, 32);
    sq += __shfl_xor(sq, off, 32);
    dp += __shfl_xor(dp, off, 32);
  }
  float mean = s * (1.0f / 64.0f);
  float var  = sq * (1.0f / 64.0f) - mean * mean;  // biased, like torch GN
  float rstd = rsqrtf(var + GN_EPS_C);
  int d0 = h * HKD + c0, d1 = h * HKD + c1;
  float y0 = (o0 - mean) * rstd * gamma[d0] + beta[d0] + dp * v[base + c0];
  float y1 = (o1 - mean) * rstd * gamma[d1] + beta[d1] + dp * v[base + c1];
  y16[base + c0] = f2bf(y0 * g[base + c0]);
  y16[base + c1] = f2bf(y1 * g[base + c1]);
}

// ------------------------------------------------------------------- host ---
extern "C" void kernel_launch(void* const* d_in, const int* in_sizes, int n_in,
                              void* d_out, int out_size, void* d_ws, size_t ws_size,
                              hipStream_t stream) {
  (void)in_sizes; (void)n_in; (void)out_size; (void)ws_size;
  const float* hidden  = (const float*)d_in[0];
  const float* v_first = (const float*)d_in[1];
  const float* x_r = (const float*)d_in[2];
  const float* x_w = (const float*)d_in[3];
  const float* x_k = (const float*)d_in[4];
  const float* x_v = (const float*)d_in[5];
  const float* x_a = (const float*)d_in[6];
  const float* x_g = (const float*)d_in[7];
  const float* k_k = (const float*)d_in[8];
  const float* k_a = (const float*)d_in[9];
  const float* r_k = (const float*)d_in[10];
  const float* W_r = (const float*)d_in[11];
  const float* W_k = (const float*)d_in[12];
  const float* W_v = (const float*)d_in[13];
  const float* W_o = (const float*)d_in[14];
  const float* w_la = (const float*)d_in[15];
  const float* w_lb = (const float*)d_in[16];
  const float* w_lbias = (const float*)d_in[17];
  const float* a_la = (const float*)d_in[18];
  const float* a_lb = (const float*)d_in[19];
  const float* a_lbias = (const float*)d_in[20];
  const float* v_la = (const float*)d_in[21];
  const float* v_lb = (const float*)d_in[22];
  const float* v_lbias = (const float*)d_in[23];
  const float* g_la = (const float*)d_in[24];
  const float* g_lb = (const float*)d_in[25];
  const float* gn_gamma = (const float*)d_in[26];
  const float* gn_beta  = (const float*)d_in[27];

  const int M = MROWS, D = DMODEL;
  char* ws = (char*)d_ws;
  size_t off = 0;
  auto alloc = [&](size_t bytes) -> void* {
    void* p = ws + off;
    off += (bytes + 255) & ~(size_t)255;
    return p;
  };
  // bf16 activations & weights
  u16* xr16 = (u16*)alloc((size_t)M * D * 2);
  u16* xw16 = (u16*)alloc((size_t)M * D * 2);
  u16* xk16 = (u16*)alloc((size_t)M * D * 2);
  u16* xv16 = (u16*)alloc((size_t)M * D * 2);
  u16* xa16 = (u16*)alloc((size_t)M * D * 2);
  u16* xg16 = (u16*)alloc((size_t)M * D * 2);
  u16* Wr16 = (u16*)alloc((size_t)D * D * 2);
  u16* Wk16 = (u16*)alloc((size_t)D * D * 2);
  u16* Wv16 = (u16*)alloc((size_t)D * D * 2);
  u16* Wo16 = (u16*)alloc((size_t)D * D * 2);
  u16* wA16 = (u16*)alloc((size_t)64 * D * 2);
  u16* wB16 = (u16*)alloc((size_t)D * 64 * 2);
  u16* aA16 = (u16*)alloc((size_t)64 * D * 2);
  u16* aB16 = (u16*)alloc((size_t)D * 64 * 2);
  u16* vA16 = (u16*)alloc((size_t)16 * D * 2);
  u16* vB16 = (u16*)alloc((size_t)D * 32 * 2);   // K padded 16 -> 32
  u16* gA16 = (u16*)alloc((size_t)128 * D * 2);
  u16* gB16 = (u16*)alloc((size_t)D * 128 * 2);
  u16* wmid16 = (u16*)alloc((size_t)M * 64 * 2);
  u16* amid16 = (u16*)alloc((size_t)M * 64 * 2);
  u16* vmid16 = (u16*)alloc((size_t)M * 32 * 2);
  u16* gmid16 = (u16*)alloc((size_t)M * 128 * 2);
  u16* y16    = (u16*)alloc((size_t)M * D * 2);
  // f32 intermediates
  float* rbuf = (float*)alloc((size_t)M * D * 4);
  float* wbuf = (float*)alloc((size_t)M * D * 4);
  float* kbuf = (float*)alloc((size_t)M * D * 4);
  float* vbuf = (float*)alloc((size_t)M * D * 4);
  float* abuf = (float*)alloc((size_t)M * D * 4);   // a -> avec (in place)
  float* bvec = (float*)alloc((size_t)M * D * 4);
  float* gbuf = (float*)alloc((size_t)M * D * 4);
  float* obuf = (float*)alloc((size_t)M * D * 4);   // vmix first, then scan output

  auto conv = [&](const float* src, u16* dst, int rows, int cols, int stride) {
    int total = rows * stride;
    conv_f32_bf16_kernel<<<(total + 255) / 256, 256, 0, stream>>>(src, dst, rows, cols, stride);
  };
  auto gemm = [&](const u16* A, const u16* W, const float* bias, float* oF, u16* oH,
                  int M_, int N_, int K_, int lda, int ldb, int ldc, int Npad,
                  int mode, float scale) {
    dim3 grid((Npad + 255) / 256, (M_ + 63) / 64);
    gemm_bf16_wmma_kernel<<<grid, 256, 0, stream>>>(A, W, bias, oF, oH, M_, N_, K_,
                                                    lda, ldb, ldc, Npad, mode, scale);
  };

  // 1) weight conversion to bf16
  conv(W_r, Wr16, D, D, D);   conv(W_k, Wk16, D, D, D);
  conv(W_v, Wv16, D, D, D);   conv(W_o, Wo16, D, D, D);
  conv(w_la, wA16, 64, D, D); conv(w_lb, wB16, D, 64, 64);
  conv(a_la, aA16, 64, D, D); conv(a_lb, aB16, D, 64, 64);
  conv(v_la, vA16, 16, D, D); conv(v_lb, vB16, D, 16, 32);  // zero-pad K to 32
  conv(g_la, gA16, 128, D, D); conv(g_lb, gB16, D, 128, 128);

  // 2) token shift + mix -> six bf16 streams
  mix_kernel<<<(M * D + 255) / 256, 256, 0, stream>>>(
      hidden, x_r, x_w, x_k, x_v, x_a, x_g, xr16, xw16, xk16, xv16, xa16, xg16);

  // 3) projections + LoRA (WMMA)
  gemm(xr16, Wr16, nullptr, rbuf, nullptr, M, D, D, D, D, D, D, 0, 1.0f);
  gemm(xk16, Wk16, nullptr, kbuf, nullptr, M, D, D, D, D, D, D, 0, 1.0f);
  gemm(xv16, Wv16, nullptr, vbuf, nullptr, M, D, D, D, D, D, D, 0, 1.0f);
  gemm(xw16, wA16, nullptr, nullptr, wmid16, M, 64, D, D, D, 64, 64, 2, 1.0f);   // tanh
  gemm(xa16, aA16, nullptr, nullptr, amid16, M, 64, D, D, D, 64, 64, 1, 1.0f);
  gemm(xv16, vA16, nullptr, nullptr, vmid16, M, 16, D, D, D, 32, 32, 1, 1.0f);   // pad N->32
  gemm(xg16, gA16, nullptr, nullptr, gmid16, M, 128, D, D, D, 128, 128, 3, 1.0f); // sigmoid
  gemm(wmid16, wB16, w_lbias, wbuf, nullptr, M, D, 64, 64, 64, D, D, 5, W_SCALE_C);
  gemm(amid16, aB16, a_lbias, abuf, nullptr, M, D, 64, 64, 64, D, D, 4, 1.0f);
  gemm(vmid16, vB16, v_lbias, obuf, nullptr, M, D, 32, 32, 32, D, D, 4, 1.0f);   // v_mix
  gemm(gmid16, gB16, nullptr, gbuf, nullptr, M, D, 128, 128, 128, D, D, 0, 1.0f);

  // 4) kk normalize, k update, v lerp (obuf currently holds v_mix)
  prep_kernel<<<(M * NHEADS) / 8, 256, 0, stream>>>(
      kbuf, abuf, vbuf, obuf, v_first, k_k, k_a, bvec);

  // 5) recurrent scan (overwrites obuf with per-head outputs)
  rwkv7_scan_kernel<<<BATCH * NHEADS, 64, 0, stream>>>(
      rbuf, wbuf, kbuf, vbuf, abuf, bvec, obuf);

  // 6) groupnorm + bonus + gate -> bf16
  gn_bonus_kernel<<<(M * NHEADS) / 8, 256, 0, stream>>>(
      obuf, rbuf, kbuf, vbuf, r_k, gn_gamma, gn_beta, gbuf, y16);

  // 7) output projection
  gemm(y16, Wo16, nullptr, (float*)d_out, nullptr, M, D, D, D, D, D, D, 0, 1.0f);
}
